// CustomLSTM_73658689126961
// MI455X (gfx1250) — matrix operand order repaired and checked by reading, not compiled, not run on hardware
//
#include <hip/hip_runtime.h>
#include <hip/hip_bf16.h>

// ---------------------------------------------------------------------------
// LSTM: B=32, S=1024, I=512, H=512  (4H = 2048)
//   Phase 1: xW = x@W + bias  -- bf16 WMMA GEMM; W^T slab staged in LDS by TDM.
//   Phase 2: persistent scan (8 blocks, 1 grid barrier/step); U^T (256KB/block)
//            staged in LDS ONCE by 4 tensor_load_to_lds ops; c in VGPRs;
//            h ping-pongs through L2 as bf16; xW prefetched one step ahead.
//   B-fragments are re-read from LDS every iteration; an asm memory clobber in
//   the GEMM's mg loop stops LICM from hoisting 512 VGPRs of loop-invariant
//   fragments (which previously spilled to scratch).
// ---------------------------------------------------------------------------

#define Bn   32
#define Sn   1024
#define In   512
#define Hn   512
#define G4   2048
#define Mrows (Bn * Sn)

typedef __attribute__((ext_vector_type(16))) __bf16 v16bf;
typedef __attribute__((ext_vector_type(8)))  float  v8f;
typedef __attribute__((ext_vector_type(4)))  unsigned int uint4v;
typedef __attribute__((ext_vector_type(4)))  unsigned int u32x4;
typedef __attribute__((ext_vector_type(8)))  int i32x8;
typedef __attribute__((ext_vector_type(4)))  int i32x4;

union FragU {                 // 32 bytes = 16 bf16 = one WMMA A/B fragment per lane
    uint4v q[2];
    v16bf  bf;
};

__device__ __forceinline__ unsigned short f32_to_bf16_rne(float f) {
    unsigned u = __builtin_bit_cast(unsigned, f);
    unsigned r = u + 0x7FFFu + ((u >> 16) & 1u);
    return (unsigned short)(r >> 16);
}

__device__ __forceinline__ float sigmoidf_fast(float x) {
    return 1.0f / (1.0f + __expf(-x));
}

// TDM: 1-D copy of 32768 bf16 elements (64KB) global -> LDS.
// Descriptor per CDNA5 ISA ch.8: group0 {count=1, lds_addr, global_addr, type=2},
// group1 {data_size=2B, tensor_dim0=tile_dim0=32768, dim1=1, stride0=32768}.
__device__ __forceinline__ void tdm_load_64k(const unsigned short* gsrc,
                                             unsigned lds_byte_addr) {
    unsigned long long ga = (unsigned long long)(const void*)gsrc;
    u32x4 g0 = { 1u,
                 lds_byte_addr,
                 (unsigned)ga,
                 (unsigned)((ga >> 32) & 0x1FFFFFFull) | (2u << 30) };
    i32x8 g1 = { 0x10000, (int)0x80000000, 0x10000, (int)0x80000000,
                 1, 32768, 0, 0 };
    i32x4 z4 = { 0, 0, 0, 0 };
#if __clang_major__ >= 23
    i32x8 z8 = { 0, 0, 0, 0, 0, 0, 0, 0 };
    __builtin_amdgcn_tensor_load_to_lds(g0, g1, z4, z4, z8, 0);
#else
    __builtin_amdgcn_tensor_load_to_lds(g0, g1, z4, z4, 0);
#endif
}

// ------------------------- workspace layout (bytes) ------------------------
#define XW_OFF   0ull
#define XB_OFF   (XW_OFF + (unsigned long long)Mrows * G4 * 4ull)
#define WT_OFF   (XB_OFF + (unsigned long long)Mrows * In * 2ull)
#define UT_OFF   (WT_OFF + (unsigned long long)G4 * In * 2ull)
#define HB_OFF   (UT_OFF + (unsigned long long)G4 * Hn * 2ull)
#define CNT_OFF  (HB_OFF + 2ull * Bn * Hn * 2ull)
#define WS_TOTAL (CNT_OFF + 64ull)

#define OUT_HSEQ 0u
#define OUT_HT   ((unsigned)(Bn * Sn * Hn))
#define OUT_CT   (OUT_HT + (unsigned)(Bn * Hn))

// ---------------------------------------------------------------------------
__global__ void lstm_init_kernel(unsigned short* __restrict__ hbuf,
                                 unsigned* __restrict__ cnt) {
    int i = blockIdx.x * blockDim.x + threadIdx.x;
    if (i == 0) *cnt = 0u;
    int stride = blockDim.x * gridDim.x;
    for (int k = i; k < 2 * Bn * Hn; k += stride) hbuf[k] = 0;
}

__global__ void lstm_cvt_bf16_kernel(const float* __restrict__ src,
                                     unsigned short* __restrict__ dst, int n) {
    int i = blockIdx.x * blockDim.x + threadIdx.x;
    int stride = blockDim.x * gridDim.x;
    for (; i < n; i += stride) dst[i] = f32_to_bf16_rne(src[i]);
}

// src: [K, N] row-major f32 -> dst: [N, K] row-major bf16
__global__ void lstm_cvt_t_bf16_kernel(const float* __restrict__ src,
                                       unsigned short* __restrict__ dst,
                                       int K, int N) {
    int i = blockIdx.x * blockDim.x + threadIdx.x;
    int stride = blockDim.x * gridDim.x;
    int total = K * N;
    for (; i < total; i += stride) {
        int n = i / K, k = i - n * K;
        dst[i] = f32_to_bf16_rne(src[k * N + n]);
    }
}

// ---------------------------------------------------------------------------
// Phase 1: xW = x @ W + bias.  Block: 8 waves; B-slab (64 cols x 512 K, 64KB)
// TDM-staged in LDS, reused by 8 waves x 8 M-groups.  A software-pipelined
// from global.  grid = (N/64, 32).
// ---------------------------------------------------------------------------
__global__ __launch_bounds__(256)
void lstm_gemm_xw_kernel(const unsigned short* __restrict__ xb,   // [32768,512]
                         const unsigned short* __restrict__ wt,   // [2048,512] W^T
                         const float* __restrict__ bias,          // [2048]
                         float* __restrict__ xw)                  // [32768,2048]
{
    __shared__ __align__(16) unsigned short lds_b[64 * In];       // 64 KB

    const int lane = threadIdx.x & 31;
    const int wave = threadIdx.x >> 5;
    const int nb   = blockIdx.x * 64;
    const int lh   = lane >> 4;
    const int ml   = lane & 15;

    if (wave == 0) {
        tdm_load_64k(wt + (unsigned long long)nb * In,
                     (unsigned)(unsigned long long)&lds_b[0]);
        __builtin_amdgcn_s_wait_tensorcnt(0);
    }
    __syncthreads();

    for (int mg = 0; mg < 8; ++mg) {
        // Block LICM from hoisting the (loop-invariant) LDS B-fragments out of
        // this loop: 64 frags = 512 VGPRs would spill to scratch.  Re-reading
        // LDS each mg iteration is the cheap path (ds_load_b128).
        asm volatile("" ::: "memory");

        const int mt = (blockIdx.y * 8 + mg) * 8 + wave;          // 0..2047
        const int m0 = mt * 16;

        v8f acc[4];
#pragma unroll
        for (int g = 0; g < 4; ++g) acc[g] = (v8f){};

        const unsigned short* arow = xb + (m0 + ml) * In + lh * 8;
        FragU a;
        a.q[0] = *(const uint4v*)(arow);
        a.q[1] = *(const uint4v*)(arow + 16);

        for (int kt = 0; kt < 16; ++kt) {
            FragU an = a;
            if (kt < 15) {                                        // prefetch next A
                const unsigned short* ap = arow + (kt + 1) * 32;
                an.q[0] = *(const uint4v*)(ap);
                an.q[1] = *(const uint4v*)(ap + 16);
            }
#pragma unroll
            for (int g = 0; g < 4; ++g) {
                FragU b;
                const int boff = (g * 16 + ml) * In + kt * 32 + lh * 8;
                b.q[0] = *(const uint4v*)(&lds_b[boff]);
                b.q[1] = *(const uint4v*)(&lds_b[boff + 16]);
                acc[g] = __builtin_amdgcn_wmma_f32_16x16x32_bf16(
                    false, a.bf, false, b.bf, (short)0, acc[g], false, false);
            }
            a = an;
        }

#pragma unroll
        for (int g = 0; g < 4; ++g) {
            const int col = nb + g * 16 + ml;
            const float bv = bias[col];
#pragma unroll
            for (int r = 0; r < 8; ++r) {
                const int row = m0 + r + 8 * lh;
                xw[(unsigned long long)row * G4 + col] = acc[g][r] + bv;
            }
        }
    }
}

// ---------------------------------------------------------------------------
// Phase 2: persistent recurrence.  grid = 8 blocks x 256 threads.
// Block bx covers h-cols [64bx, 64bx+64): wave = (mtile = w>>2, hgi = w&3).
// Its U^T working set (4 gates x 64 cols x 512 K = 256 KB) is TDM-loaded into
// LDS once; inner loop feeds all B-fragments from LDS.  c stays in VGPRs.
// The per-step barrier (__syncthreads + atomics) already blocks cross-step
// hoisting of the LDS reads.
// ---------------------------------------------------------------------------
__global__ __launch_bounds__(256)
void lstm_scan_kernel(const float* __restrict__ xw,              // [32768,2048]
                      const unsigned short* __restrict__ ut,     // [2048,512] U^T
                      unsigned short* __restrict__ hbuf,         // [2][32][512]
                      float* __restrict__ out,
                      unsigned* __restrict__ cnt)
{
    __shared__ __align__(16) unsigned short lds_u[256 * Hn];     // 256 KB

    const int lane  = threadIdx.x & 31;
    const int wave  = threadIdx.x >> 5;
    const int bx    = blockIdx.x;            // 0..7
    const int mtile = wave >> 2;             // 0..1
    const int hgi   = wave & 3;              // 0..3
    const int hg    = bx * 4 + hgi;          // 0..31
    const int lh    = lane >> 4;
    const int ml    = lane & 15;
    const int m0    = mtile * 16;
    const int jcol  = hg * 16 + ml;
    const unsigned nblocks = gridDim.x;

    // One 64KB TDM descriptor per gate: ut rows [g*512 + 64bx, +64) (contiguous).
    if (wave == 0) {
        const unsigned lds_base = (unsigned)(unsigned long long)&lds_u[0];
#pragma unroll
        for (int g = 0; g < 4; ++g)
            tdm_load_64k(ut + (unsigned long long)(g * Hn + bx * 64) * Hn,
                         lds_base + (unsigned)g * 65536u);
        __builtin_amdgcn_s_wait_tensorcnt(0);
    }
    __syncthreads();

    const int cbase[4] = { hg * 16, Hn + hg * 16, 2 * Hn + hg * 16, 3 * Hn + hg * 16 };
    const int lbase[4] = { (0 * 64 + hgi * 16 + ml) * Hn, (1 * 64 + hgi * 16 + ml) * Hn,
                           (2 * 64 + hgi * 16 + ml) * Hn, (3 * 64 + hgi * 16 + ml) * Hn };

    float cst[8];
#pragma unroll
    for (int r = 0; r < 8; ++r) cst[r] = 0.0f;

    for (int t = 0; t < Sn; ++t) {
        const unsigned short* hp = hbuf + (t & 1) * (Bn * Hn);
        unsigned short*       hn = hbuf + ((t + 1) & 1) * (Bn * Hn);

        // accumulators start at xW[b*S + t, gate_col] (bias folded in)
        v8f acc[4];
#pragma unroll
        for (int g = 0; g < 4; ++g)
#pragma unroll
            for (int r = 0; r < 8; ++r)
                acc[g][r] = xw[(unsigned long long)((m0 + r + 8 * lh) * Sn + t) * G4
                               + cbase[g] + ml];

        // hide next step's xW latency behind this step's compute
        if (t + 1 < Sn) {
#pragma unroll
            for (int g = 0; g < 4; ++g)
#pragma unroll
                for (int r = 0; r < 8; ++r)
                    __builtin_prefetch(
                        &xw[(unsigned long long)((m0 + r + 8 * lh) * Sn + t + 1) * G4
                            + cbase[g] + ml], 0, 3);
        }

        // gates += h_t @ U   (K = 512 -> 16 k-steps x 4 WMMA; B from LDS)
        for (int kt = 0; kt < 16; ++kt) {
            FragU a;
            const unsigned short* ap = hp + (m0 + ml) * Hn + kt * 32 + lh * 8;
            a.q[0] = *(const uint4v*)(ap);
            a.q[1] = *(const uint4v*)(ap + 16);
#pragma unroll
            for (int g = 0; g < 4; ++g) {
                FragU b;
                const int boff = lbase[g] + kt * 32 + lh * 8;
                b.q[0] = *(const uint4v*)(&lds_u[boff]);
                b.q[1] = *(const uint4v*)(&lds_u[boff + 16]);
                acc[g] = __builtin_amdgcn_wmma_f32_16x16x32_bf16(
                    false, a.bf, false, b.bf, (short)0, acc[g], false, false);
            }
        }

        // pointwise gates; c in registers; h -> out (f32) and hbuf (bf16)
#pragma unroll
        for (int r = 0; r < 8; ++r) {
            const int b = m0 + r + 8 * lh;
            const float iv = sigmoidf_fast(acc[0][r]);
            const float fv = sigmoidf_fast(acc[1][r]);
            const float gv = tanhf(acc[2][r]);
            const float ov = sigmoidf_fast(acc[3][r]);
            const float cn = fv * cst[r] + iv * gv;
            cst[r] = cn;
            const float hv = ov * tanhf(cn);
            out[OUT_HSEQ + (unsigned)b * (Sn * Hn) + (unsigned)t * Hn + jcol] = hv;
            hn[b * Hn + jcol] = f32_to_bf16_rne(hv);
            if (t == Sn - 1) {
                out[OUT_HT + (unsigned)b * Hn + jcol] = hv;
                out[OUT_CT + (unsigned)b * Hn + jcol] = cn;
            }
        }

        // device-wide barrier (8 resident blocks): release, arrive, spin, acquire
        __threadfence();
        __syncthreads();
        if (threadIdx.x == 0) {
            atomicAdd(cnt, 1u);
            const unsigned target = (unsigned)(t + 1) * nblocks;
            while (__hip_atomic_load(cnt, __ATOMIC_ACQUIRE, __HIP_MEMORY_SCOPE_AGENT)
                   < target)
                __builtin_amdgcn_s_sleep(2);
        }
        __syncthreads();
    }
}

// ---------------------------------------------------------------------------
extern "C" void kernel_launch(void* const* d_in, const int* in_sizes, int n_in,
                              void* d_out, int out_size, void* d_ws, size_t ws_size,
                              hipStream_t stream) {
    (void)in_sizes; (void)n_in; (void)out_size;
    if (ws_size < WS_TOTAL) return;

    const float* x    = (const float*)d_in[0];
    const float* W    = (const float*)d_in[1];
    const float* U    = (const float*)d_in[2];
    const float* bias = (const float*)d_in[3];
    float* out = (float*)d_out;

    char* ws = (char*)d_ws;
    float*          xw   = (float*)(ws + XW_OFF);
    unsigned short* xb   = (unsigned short*)(ws + XB_OFF);
    unsigned short* wt   = (unsigned short*)(ws + WT_OFF);
    unsigned short* ut   = (unsigned short*)(ws + UT_OFF);
    unsigned short* hbuf = (unsigned short*)(ws + HB_OFF);
    unsigned*       cnt  = (unsigned*)(ws + CNT_OFF);

    lstm_init_kernel<<<8, 256, 0, stream>>>(hbuf, cnt);
    lstm_cvt_bf16_kernel<<<2048, 256, 0, stream>>>(x, xb, Mrows * In);
    lstm_cvt_t_bf16_kernel<<<512, 256, 0, stream>>>(W, wt, In, G4);
    lstm_cvt_t_bf16_kernel<<<512, 256, 0, stream>>>(U, ut, Hn, G4);

    dim3 g1(G4 / 64, 32);                 // each block: 8 waves x 8 M-groups
    lstm_gemm_xw_kernel<<<g1, 256, 0, stream>>>(xb, wt, bias, xw);

    lstm_scan_kernel<<<8, 256, 0, stream>>>(xw, ut, hbuf, out, cnt);
}
